// OutputModule_55568286876197
// MI455X (gfx1250) — compile-verified, matrix operand order checked by reading, not run on hardware
//
#include <hip/hip_runtime.h>
#include <hip/hip_bf16.h>
#include <stdint.h>

// ---------------------------------------------------------------------------
// Types for CDNA5 WMMA (wave32, 16x16x32 bf16 -> f32)
// ---------------------------------------------------------------------------
typedef __attribute__((ext_vector_type(16))) __bf16 v16bf;
typedef __attribute__((ext_vector_type(8)))  float  v8f;

union Frag {
    v16bf v;
    uint4 q[2];
};

// ---------------------------------------------------------------------------
// Helpers
// ---------------------------------------------------------------------------
__device__ __forceinline__ unsigned short f2bf(float f) {
    unsigned int u = __float_as_uint(f);
    u += 0x7FFFu + ((u >> 16) & 1u);   // round-to-nearest-even
    return (unsigned short)(u >> 16);
}
__device__ __forceinline__ float bf2f(unsigned short u) {
    return __uint_as_float(((unsigned int)u) << 16);
}
__device__ __forceinline__ float silu(float x) {
    return x / (1.0f + __expf(-x));
}

#define EMBED    512
#define HIDDEN   1024
#define IN_DIM   1536
#define N_GAUSS  50
#define N_GRAPHS 64

// ---------------------------------------------------------------------------
// Kernel: zero the output buffer (energy 64 + forces 30000, fp32)
// ---------------------------------------------------------------------------
__global__ void zero_f32(float* __restrict__ p, int n) {
    int i = blockIdx.x * blockDim.x + threadIdx.x;
    if (i < n) p[i] = 0.0f;
}

// ---------------------------------------------------------------------------
// Kernel: convert fp32 W[K,N] -> bf16 W^T[N,K]  (B-operand staging layout)
// ---------------------------------------------------------------------------
__global__ void conv_transpose_bf16(const float* __restrict__ in,
                                    unsigned short* __restrict__ out,
                                    int K, int N) {
    long long tid = (long long)blockIdx.x * blockDim.x + threadIdx.x;
    long long tot = (long long)K * N;
    if (tid >= tot) return;
    int n = (int)(tid / K);
    int k = (int)(tid % K);
    out[tid] = f2bf(in[(size_t)k * N + n]);
}

// ---------------------------------------------------------------------------
// Kernel: build bf16 edge-input [Mc, 1536] = [x[src] | x[dst] | rbf(dist)]
// One block (256 threads) per edge.
// ---------------------------------------------------------------------------
__global__ __launch_bounds__(256)
void build_input(const float* __restrict__ x,
                 const int* __restrict__ edge_index,
                 const float* __restrict__ dist,
                 const float* __restrict__ rbfW,   // [50,512]
                 const float* __restrict__ rbfb,   // [512]
                 unsigned short* __restrict__ in_bf,
                 int c0, int E_total, int Mc) {
    int e = blockIdx.x;
    if (e >= Mc) return;
    int ge   = c0 + e;
    int srcn = edge_index[ge];
    int dstn = edge_index[E_total + ge];

    __shared__ float gs[64];
    int t = threadIdx.x;
    const float delta = 12.0f / 49.0f;
    const float coeff = -0.5f / (delta * delta);
    if (t < N_GAUSS) {
        float d = dist[ge];
        float u = d - delta * (float)t;
        gs[t] = __expf(coeff * u * u);
    }
    __syncthreads();

    unsigned short* row = in_bf + (size_t)e * IN_DIM;
    const float* xs = x + (size_t)srcn * EMBED;
    const float* xd = x + (size_t)dstn * EMBED;
    for (int j = t; j < EMBED; j += 256) {
        row[j]         = f2bf(xs[j]);
        row[EMBED + j] = f2bf(xd[j]);
        float acc = rbfb[j];
#pragma unroll 10
        for (int g = 0; g < N_GAUSS; ++g)
            acc += gs[g] * rbfW[g * EMBED + j];
        row[2 * EMBED + j] = f2bf(acc);
    }
}

// ---------------------------------------------------------------------------
// Kernel: bf16 WMMA GEMM with fused bias + SiLU (+ optional residual).
//   Out[M,1024] = maybe(Res) + silu(A[M,K] @ W + bias)      (bf16 out)
//   Bt = W^T stored [1024, K] bf16.
//
// Block tile 128(M) x 256(N), 256 threads = 8 waves arranged 2(M) x 4(N);
// each wave computes a 64x64 sub-tile = 4x4 v_wmma_f32_16x16x32_bf16 accs
// (16 WMMA per K-step, 16 ds_load_b128 fragment loads -> 1 load/WMMA).
//
// Staging uses CDNA5 async loads: global_load_async_to_lds_b128 writes LDS
// directly (ASYNCcnt), double-buffered so tile k+1 streams in during the
// WMMAs of tile k with a single workgroup barrier per K-step.
// ---------------------------------------------------------------------------
#define BM 128
#define BN 256
#define BK 32

__global__ __launch_bounds__(256)
void gemm_bf16_silu(const unsigned short* __restrict__ A,    // [M,K] bf16
                    const unsigned short* __restrict__ Bt,   // [1024,K] bf16
                    const float* __restrict__ bias,          // [1024]
                    const unsigned short* __restrict__ Res,  // [M,1024] bf16 or null
                    unsigned short* __restrict__ Out,        // [M,1024] bf16
                    int M, int K) {
    // padded LDS rows: stride 40 ushorts (80B) -> conflict-free b128, 16B aligned
    __shared__ unsigned short As[2][BM][40];   // 2 x 10240 B
    __shared__ unsigned short Bs[2][BN][40];   // 2 x 20480 B

    const int tid  = threadIdx.x;
    const int bn   = blockIdx.x;          // 0..3  (N / 256)
    const int bm   = blockIdx.y;
    const int m0   = bm * BM;
    const int n0   = bn * BN;
    const int wave = tid >> 5;
    const int lane = tid & 31;
    const int wm   = wave & 1;            // 2 row-groups of 64
    const int wn   = wave >> 1;           // 4 col-groups of 64
    const int hl   = lane >> 4;           // half-wave select
    const int l16  = lane & 15;

    v8f acc[4][4];
#pragma unroll
    for (int a = 0; a < 4; ++a)
#pragma unroll
        for (int b = 0; b < 4; ++b)
#pragma unroll
            for (int i = 0; i < 8; ++i) acc[a][b][i] = 0.0f;

    const int nKT = K / BK;

    // issue CDNA5 async global->LDS loads for K-tile kt into buffer buf
    auto issue_tile = [&](int kt, int buf) {
        // A tile: 128 rows x 32 cols bf16 = 512 x 16B chunks, 2 per thread
#pragma unroll
        for (int i = 0; i < 2; ++i) {
            int li   = tid + i * 256;
            int row  = li >> 2;
            int c8   = (li & 3) * 8;
            int arow = m0 + row; if (arow >= M) arow = M - 1;  // clamp, discarded later
            unsigned lds = (unsigned)(uintptr_t)&As[buf][row][c8];
            const void* g = (const void*)&A[(size_t)arow * K + (size_t)kt * BK + c8];
            asm volatile("global_load_async_to_lds_b128 %0, %1, off"
                         :: "v"(lds), "v"(g) : "memory");
        }
        // B tile: 256 rows x 32 cols bf16 = 1024 x 16B chunks, 4 per thread
#pragma unroll
        for (int i = 0; i < 4; ++i) {
            int li  = tid + i * 256;
            int row = li >> 2;
            int c8  = (li & 3) * 8;
            int brow = n0 + row;
            unsigned lds = (unsigned)(uintptr_t)&Bs[buf][row][c8];
            const void* g = (const void*)&Bt[(size_t)brow * K + (size_t)kt * BK + c8];
            asm volatile("global_load_async_to_lds_b128 %0, %1, off"
                         :: "v"(lds), "v"(g) : "memory");
        }
    };

    issue_tile(0, 0);

    for (int kt = 0; kt < nKT; ++kt) {
        const int cur = kt & 1;
        // my async loads for tile kt have landed in LDS:
        asm volatile("s_wait_asynccnt 0" ::: "memory");
        // everyone's landed, and everyone is done reading buffer (kt+1)&1:
        __syncthreads();
        if (kt + 1 < nKT) issue_tile(kt + 1, cur ^ 1);  // streams during WMMAs below

        // fragment loads per CDNA5 16-bit 16x32 layout:
        //   lane(0-15): q0 = K[0..7],  q1 = K[16..23]
        //   lane(16-31):q0 = K[8..15], q1 = K[24..31]
        Frag af[4], bfr[4];
#pragma unroll
        for (int a = 0; a < 4; ++a) {
            int r = wm * 64 + a * 16 + l16;
            af[a].q[0] = *(const uint4*)&As[cur][r][hl * 8];
            af[a].q[1] = *(const uint4*)&As[cur][r][16 + hl * 8];
        }
#pragma unroll
        for (int b = 0; b < 4; ++b) {
            int c = wn * 64 + b * 16 + l16;
            bfr[b].q[0] = *(const uint4*)&Bs[cur][c][hl * 8];
            bfr[b].q[1] = *(const uint4*)&Bs[cur][c][16 + hl * 8];
        }
#pragma unroll
        for (int a = 0; a < 4; ++a)
#pragma unroll
            for (int b = 0; b < 4; ++b)
                acc[a][b] = __builtin_amdgcn_wmma_f32_16x16x32_bf16(
                    false, af[a].v, false, bfr[b].v,
                    (short)0, acc[a][b], false, false);
    }

    // epilogue: bias + SiLU (+ residual), bf16 store
#pragma unroll
    for (int a = 0; a < 4; ++a) {
#pragma unroll
        for (int b = 0; b < 4; ++b) {
            int col = n0 + wn * 64 + b * 16 + l16;
            float bv = bias[col];
#pragma unroll
            for (int i = 0; i < 8; ++i) {
                int row = m0 + wm * 64 + a * 16 + hl * 8 + i;
                if (row < M) {
                    float v = silu(acc[a][b][i] + bv);
                    size_t o = (size_t)row * HIDDEN + col;
                    if (Res) v += bf2f(Res[o]);
                    Out[o] = f2bf(v);
                }
            }
        }
    }
}

// ---------------------------------------------------------------------------
// Kernel: output head GEMV  pair[e] = h[e,:] . w + b   (one wave per edge)
// ---------------------------------------------------------------------------
__global__ __launch_bounds__(256)
void gemv_out(const unsigned short* __restrict__ h,   // [Mc,1024] bf16
              const float* __restrict__ w,            // [1024]
              const float* __restrict__ b,            // [1]
              float* __restrict__ pair, int Mc) {
    int wv   = threadIdx.x >> 5;
    int lane = threadIdx.x & 31;
    int e    = blockIdx.x * 8 + wv;
    if (e >= Mc) return;
    const unsigned short* hr = h + (size_t)e * HIDDEN;
    float s = 0.0f;
#pragma unroll 8
    for (int k = lane; k < HIDDEN; k += 32) s += bf2f(hr[k]) * w[k];
#pragma unroll
    for (int m = 16; m >= 1; m >>= 1) s += __shfl_xor(s, m, 32);
    if (lane == 0) pair[e] = s + b[0];
}

// ---------------------------------------------------------------------------
// Kernel: energy segment-sum into 64 graph bins (LDS bins -> global atomics)
// ---------------------------------------------------------------------------
__global__ __launch_bounds__(256)
void reduce_energy(const float* __restrict__ epair,
                   const int* __restrict__ src,     // chunk-offset src ids
                   const int* __restrict__ batch,
                   float* __restrict__ out_energy,  // [64]
                   int Mc, float scale) {
    __shared__ float bins[N_GRAPHS];
    int t = threadIdx.x;
    if (t < N_GRAPHS) bins[t] = 0.0f;
    __syncthreads();
    for (int e = blockIdx.x * blockDim.x + t; e < Mc; e += gridDim.x * blockDim.x) {
        atomicAdd(&bins[batch[src[e]]], epair[e]);
    }
    __syncthreads();
    if (t < N_GRAPHS) atomicAdd(&out_energy[t], bins[t] * scale);
}

// ---------------------------------------------------------------------------
// Kernel: force scatter-add  forces[src[e],:] += fpair[e]*vec_hat[e,:]/32
// ---------------------------------------------------------------------------
__global__ __launch_bounds__(256)
void reduce_force(const float* __restrict__ fpair,
                  const float* __restrict__ vhat,   // chunk-offset [Mc,3]
                  const int* __restrict__ src,      // chunk-offset
                  float* __restrict__ out_forces,   // [N_NODES,3]
                  int Mc, float inv_conn) {
    int e = blockIdx.x * blockDim.x + threadIdx.x;
    if (e >= Mc) return;
    float s = fpair[e] * inv_conn;
    int n = src[e];
    atomicAdd(&out_forces[(size_t)n * 3 + 0], s * vhat[(size_t)e * 3 + 0]);
    atomicAdd(&out_forces[(size_t)n * 3 + 1], s * vhat[(size_t)e * 3 + 1]);
    atomicAdd(&out_forces[(size_t)n * 3 + 2], s * vhat[(size_t)e * 3 + 2]);
}

// ---------------------------------------------------------------------------
// Host orchestration
// ---------------------------------------------------------------------------
#define CHUNK 12800

extern "C" void kernel_launch(void* const* d_in, const int* in_sizes, int n_in,
                              void* d_out, int out_size, void* d_ws, size_t ws_size,
                              hipStream_t stream) {
    const float* x        = (const float*)d_in[0];
    const int*   edge_idx = (const int*)  d_in[1];
    const int*   batch    = (const int*)  d_in[2];
    const float* dist     = (const float*)d_in[3];
    const float* vec_hat  = (const float*)d_in[4];
    const float* rbf_W    = (const float*)d_in[5];
    const float* rbf_b    = (const float*)d_in[6];
    const float* e_Win    = (const float*)d_in[7];
    const float* e_bin    = (const float*)d_in[8];
    const float* e_Wh     = (const float*)d_in[9];
    const float* e_bh     = (const float*)d_in[10];
    const float* e_Wout   = (const float*)d_in[11];
    const float* e_bout   = (const float*)d_in[12];
    const float* f_Win    = (const float*)d_in[13];
    const float* f_bin    = (const float*)d_in[14];
    const float* f_Wh     = (const float*)d_in[15];
    const float* f_bh     = (const float*)d_in[16];
    const float* f_Wout   = (const float*)d_in[17];
    const float* f_bout   = (const float*)d_in[18];

    const int E = in_sizes[3];          // 100000 edges

    // ---- workspace carve-out ----
    char* ws = (char*)d_ws;
    size_t off = 0;
    auto carve = [&](size_t bytes) -> void* {
        void* p = ws + off;
        off = (off + bytes + 255) & ~(size_t)255;
        return p;
    };
    unsigned short* WtInE = (unsigned short*)carve((size_t)HIDDEN * IN_DIM * 2);
    unsigned short* WtInF = (unsigned short*)carve((size_t)HIDDEN * IN_DIM * 2);
    unsigned short* WtHE  = (unsigned short*)carve((size_t)3 * HIDDEN * HIDDEN * 2);
    unsigned short* WtHF  = (unsigned short*)carve((size_t)3 * HIDDEN * HIDDEN * 2);
    unsigned short* in_bf = (unsigned short*)carve((size_t)CHUNK * IN_DIM * 2);
    unsigned short* hA    = (unsigned short*)carve((size_t)CHUNK * HIDDEN * 2);
    unsigned short* hB    = (unsigned short*)carve((size_t)CHUNK * HIDDEN * 2);
    float*          epair = (float*)carve((size_t)CHUNK * 4);
    float*          fpair = (float*)carve((size_t)CHUNK * 4);

    // ---- one-time (per call) weight convert+transpose, output zeroing ----
    auto convT = [&](const float* src_w, unsigned short* dst_w, int K, int N) {
        long long tot = (long long)K * N;
        conv_transpose_bf16<<<(unsigned)((tot + 255) / 256), 256, 0, stream>>>(src_w, dst_w, K, N);
    };
    convT(e_Win, WtInE, IN_DIM, HIDDEN);
    convT(f_Win, WtInF, IN_DIM, HIDDEN);
    for (int i = 0; i < 3; ++i) {
        convT(e_Wh + (size_t)i * HIDDEN * HIDDEN, WtHE + (size_t)i * HIDDEN * HIDDEN, HIDDEN, HIDDEN);
        convT(f_Wh + (size_t)i * HIDDEN * HIDDEN, WtHF + (size_t)i * HIDDEN * HIDDEN, HIDDEN, HIDDEN);
    }
    zero_f32<<<(out_size + 255) / 256, 256, 0, stream>>>((float*)d_out, out_size);

    float* out_energy = (float*)d_out;        // [64]
    float* out_forces = (float*)d_out + 64;   // [10000,3]
    const float e_scale  = 1.0f / (60.0f * 32.0f);
    const float inv_conn = 1.0f / 32.0f;

    // ---- chunked edge pipeline ----
    for (int c0 = 0; c0 < E; c0 += CHUNK) {
        int Mc = E - c0; if (Mc > CHUNK) Mc = CHUNK;

        build_input<<<Mc, 256, 0, stream>>>(x, edge_idx, dist, rbf_W, rbf_b,
                                            in_bf, c0, E, Mc);

        dim3 gg(HIDDEN / BN, (unsigned)((Mc + BM - 1) / BM));
        auto run_mlp = [&](const unsigned short* WtIn, const float* bin,
                           const unsigned short* WtH, const float* bh,
                           const float* Wout, const float* bout, float* pair) {
            gemm_bf16_silu<<<gg, 256, 0, stream>>>(in_bf, WtIn, bin, nullptr, hA, Mc, IN_DIM);
            gemm_bf16_silu<<<gg, 256, 0, stream>>>(hA, WtH,                       bh,            hA, hB, Mc, HIDDEN);
            gemm_bf16_silu<<<gg, 256, 0, stream>>>(hB, WtH + 1 * HIDDEN * HIDDEN, bh + HIDDEN,   hB, hA, Mc, HIDDEN);
            gemm_bf16_silu<<<gg, 256, 0, stream>>>(hA, WtH + 2 * HIDDEN * HIDDEN, bh + 2*HIDDEN, hA, hB, Mc, HIDDEN);
            gemv_out<<<(Mc + 7) / 8, 256, 0, stream>>>(hB, Wout, bout, pair, Mc);
        };
        run_mlp(WtInE, e_bin, WtHE, e_bh, e_Wout, e_bout, epair);
        run_mlp(WtInF, f_bin, WtHF, f_bh, f_Wout, f_bout, fpair);

        reduce_energy<<<48, 256, 0, stream>>>(epair, edge_idx + c0, batch,
                                              out_energy, Mc, e_scale);
        reduce_force<<<(Mc + 255) / 256, 256, 0, stream>>>(fpair,
                                              vec_hat + (size_t)c0 * 3,
                                              edge_idx + c0, out_forces,
                                              Mc, inv_conn);
    }
}
